// TripleAttentionLayer_14843406975566
// MI455X (gfx1250) — compile-verified, matrix operand order checked by reading, not run on hardware
//
#include <hip/hip_runtime.h>
#include <hip/hip_bf16.h>
#include <math.h>

typedef __attribute__((ext_vector_type(16))) __bf16 v16bf;
typedef __attribute__((ext_vector_type(8)))  __bf16 v8bf;
typedef __attribute__((ext_vector_type(8)))  float  v8f;

#define Bn 16
#define Cc 128
#define Hh 32
#define Ww 512

#define XPITCH 136   // bf16 pitch of transposed x tile: 272 B, 16B-aligned runs
#define NWELEM (3 * Cc * 64)   // 24576 bf16 weights, [kx][co][ci]

__device__ __forceinline__ float atomicMaxF(float* addr, float val) {
    int* ai = (int*)addr;
    int old = __float_as_int(*addr);
    while (__int_as_float(old) < val) {
        int assumed = old;
        old = atomicCAS(ai, assumed, __float_as_int(val));
        if (old == assumed) break;
    }
    return __int_as_float(old);
}

__device__ __forceinline__ v16bf join16(v8bf lo, v8bf hi) {
    return __builtin_shufflevector(lo, hi, 0, 1, 2, 3, 4, 5, 6, 7,
                                           8, 9, 10, 11, 12, 13, 14, 15);
}

// ---------------- K0: one-time weight convert + reorder --------------------
// wg[co][ci][0][kx] fp32  ->  wbf[kx][co*64+ci] bf16 (49 KB, stays L2-hot)
__global__ __launch_bounds__(256)
void prep_w(const float* __restrict__ wg, __bf16* __restrict__ wbf) {
    int d = blockIdx.x * 256 + threadIdx.x;
    if (d >= NWELEM) return;
    int kx = d >> 13;            // d / 8192
    int rc = d & 8191;           // co*64 + ci
    wbf[d] = (__bf16)wg[rc * 3 + kx];
}

// ---------------- K1: grouped (1x3) conv, groups=2, via bf16 WMMA ----------
// Tap-decomposed: Y += W_kx[64co x 64ci] * Xshift_kx[64ci x 64w], kx = 0..2.
// A fragments from alds[kx][co][ci]; B fragments from transposed xlds[j][c];
// both are two contiguous 8x bf16 runs per lane -> 2x ds_load_b128 each.
__global__ __launch_bounds__(256)
void conv_wmma(const float* __restrict__ x, const __bf16* __restrict__ wbf,
               float* __restrict__ y) {
    __shared__ __bf16 alds[NWELEM];          // [kx][co*64+ci], 48 KB
    __shared__ __bf16 xlds[66 * XPITCH];     // [j][c], j = w halo, 17.9 KB

    const int b  = blockIdx.z;
    const int h  = blockIdx.y;
    const int w0 = blockIdx.x * 64;
    const int tid  = threadIdx.x;
    const int lane = tid & 31;
    const int wv   = tid >> 5;               // 8 waves
    const int m    = lane & 15;

    // Stage pre-converted weights: pure aligned vector copy (12 iters/thread).
    {
        const v8bf* src = (const v8bf*)wbf;
        v8bf* dst = (v8bf*)alds;
#pragma unroll
        for (int t = 0; t < NWELEM / 8 / 256; ++t)
            dst[t * 256 + tid] = src[t * 256 + tid];
    }
    // Stage x tile transposed: xlds[j][c] = x[b, c, h, w0-1+j].
    // 2-way unrolled so two independent loads are outstanding per wait.
    {
        const float* xrow = x + (((size_t)b * Cc) * Hh + h) * Ww;
        for (int t = tid; t < Cc * 66; t += 512) {
            int c0 = t / 66, j0 = t - c0 * 66;
            int ws0 = w0 - 1 + j0;
            float v0 = ((unsigned)ws0 < (unsigned)Ww)
                         ? xrow[(size_t)c0 * (Hh * Ww) + ws0] : 0.f;
            int t1 = t + 256;
            int c1 = t1 / 66, j1 = t1 - c1 * 66;
            int ws1 = w0 - 1 + j1;
            float v1 = (t1 < Cc * 66 && (unsigned)ws1 < (unsigned)Ww)
                         ? xrow[(size_t)c1 * (Hh * Ww) + ws1] : 0.f;
            xlds[j0 * XPITCH + c0] = (__bf16)v0;
            if (t1 < Cc * 66) xlds[j1 * XPITCH + c1] = (__bf16)v1;
        }
    }
    __syncthreads();

    const int cobase = wv * 16;              // this wave's 16 output channels
    const int g64    = (cobase >> 6) << 6;   // input-channel base of its group
    const int half   = (lane & 16) ? 8 : 0;  // K offset for hi lane half

    v8f acc[4];
#pragma unroll
    for (int i = 0; i < 4; ++i)
#pragma unroll
        for (int j = 0; j < 8; ++j) acc[i][j] = 0.f;

#pragma unroll
    for (int kx = 0; kx < 3; ++kx) {
#pragma unroll
        for (int kb = 0; kb < 2; ++kb) {
            const int K0 = kb * 32 + half;
            const __bf16* ab = alds + kx * (Cc * 64) + (cobase + m) * 64 + K0;
            v16bf a = join16(*(const v8bf*)ab, *(const v8bf*)(ab + 16));
#pragma unroll
            for (int nt = 0; nt < 4; ++nt) {
                int j = nt * 16 + m + kx;    // halo row = w_local + kx
                const __bf16* bb = xlds + j * XPITCH + g64 + K0;
                v16bf bf = join16(*(const v8bf*)bb, *(const v8bf*)(bb + 16));
                acc[nt] = __builtin_amdgcn_wmma_f32_16x16x32_bf16(
                    false, a, false, bf, (short)0, acc[nt], false, false);
            }
        }
    }

    // C/D layout: VGPR r -> M = r + (lane>=16 ? 8 : 0), N = lane&15
    const int rowoff = (lane & 16) ? 8 : 0;
#pragma unroll
    for (int nt = 0; nt < 4; ++nt) {
        int w = w0 + nt * 16 + m;
#pragma unroll
        for (int r = 0; r < 8; ++r) {
            int cco = cobase + r + rowoff;
            y[(((size_t)b * Cc + cco) * Hh + h) * Ww + w] = acc[nt][r];
        }
    }
}

// ---------------- init pool3 (max=-inf, sum=0) -----------------------------
__global__ void init_p3(float* __restrict__ p3) {
    int i = blockIdx.x * 256 + threadIdx.x;
    if (i >= Bn * 2 * Hh * Cc) return;
    int loc = i & (2 * Hh * Cc - 1);
    p3[i] = (loc < Hh * Cc) ? -3.4e38f : 0.f;
}

// ---------------- K2: all three channel pools in one pass over y -----------
// block = (b, 32-w chunk); covers all C and all H -> pool1/pool2 complete.
__global__ __launch_bounds__(256)
void pools(const float* __restrict__ y, float* __restrict__ p1,
           float* __restrict__ p2, float* __restrict__ p3) {
    const int b  = blockIdx.y;
    const int w0 = blockIdx.x * 32;
    const int tid = threadIdx.x;
    const int w  = tid & 31;      // lane == w-in-chunk
    const int hg = tid >> 5;      // 8 groups of 4 h each

    __shared__ float p2m[256], p2s[256];

    float p1mx[4], p1sm[4];
#pragma unroll
    for (int i = 0; i < 4; ++i) { p1mx[i] = -3.4e38f; p1sm[i] = 0.f; }

    for (int c = 0; c < Cc; ++c) {
        float v[4];
#pragma unroll
        for (int i = 0; i < 4; ++i) {
            int h = hg * 4 + i;
            v[i] = y[(((size_t)b * Cc + c) * Hh + h) * Ww + w0 + w];
            p1mx[i] = fmaxf(p1mx[i], v[i]);
            p1sm[i] += v[i];
        }
        // pool2: reduce over H at fixed (c, w)
        float lm = fmaxf(fmaxf(v[0], v[1]), fmaxf(v[2], v[3]));
        float ls = v[0] + v[1] + v[2] + v[3];
        p2m[tid] = lm; p2s[tid] = ls;
        __syncthreads();
        if (hg == 0) {
            float mm = p2m[w], ss = p2s[w];
#pragma unroll
            for (int j = 1; j < 8; ++j) {
                mm = fmaxf(mm, p2m[j * 32 + w]);
                ss += p2s[j * 32 + w];
            }
            size_t base = (size_t)b * 2 * Cc * Ww;
            p2[base + (size_t)c * Ww + w0 + w]           = mm;
            p2[base + Cc * Ww + (size_t)c * Ww + w0 + w] = ss * (1.f / Hh);
        }
        __syncthreads();
        // pool3: reduce over W at fixed (h, c); wave = 32 consecutive w
#pragma unroll
        for (int i = 0; i < 4; ++i) {
            float mm = v[i], ss = v[i];
            for (int off = 16; off; off >>= 1) {
                mm = fmaxf(mm, __shfl_xor(mm, off, 32));
                ss += __shfl_xor(ss, off, 32);
            }
            if (w == 0) {
                int h = hg * 4 + i;
                size_t base = (size_t)b * 2 * Hh * Cc;
                atomicMaxF(&p3[base + h * Cc + c], mm);
                atomicAdd(&p3[base + Hh * Cc + h * Cc + c], ss);
            }
        }
    }
    // pool1: reduce over C at fixed (h, w) — complete, direct store
#pragma unroll
    for (int i = 0; i < 4; ++i) {
        int h = hg * 4 + i;
        size_t base = (size_t)b * 2 * Hh * Ww;
        p1[base + (size_t)h * Ww + w0 + w]           = p1mx[i];
        p1[base + Hh * Ww + (size_t)h * Ww + w0 + w] = p1sm[i] * (1.f / Cc);
    }
}

// ---------------- K3: 7x7 conv (pad 3) over 2-channel pooled map + sigmoid -
__global__ __launch_bounds__(256)
void conv7_sig(const float* __restrict__ in, const float* __restrict__ w7,
               float* __restrict__ outp, int R, int S, float ch1scale) {
    int i = blockIdx.x * 256 + threadIdx.x;
    int total = Bn * R * S;
    if (i >= total) return;
    int s = i % S;
    int r = (i / S) % R;
    int b = i / (S * R);
    const float* in0 = in + (size_t)b * 2 * R * S;
    float gacc = 0.f;
#pragma unroll
    for (int ch = 0; ch < 2; ++ch) {
        float sc = ch ? ch1scale : 1.f;
        const float* p = in0 + (size_t)ch * R * S;
        for (int dr = 0; dr < 7; ++dr) {
            int rr = r + dr - 3;
            if ((unsigned)rr >= (unsigned)R) continue;
            for (int dc = 0; dc < 7; ++dc) {
                int ss = s + dc - 3;
                if ((unsigned)ss >= (unsigned)S) continue;
                gacc += sc * p[(size_t)rr * S + ss] * w7[(ch * 7 + dr) * 7 + dc];
            }
        }
    }
    outp[i] = 1.f / (1.f + __expf(-gacc));
}

// ---------------- K4: out = y * (s1 + s2 + s3) / 3  (in place, float4) -----
__global__ __launch_bounds__(256)
void finalk(float* __restrict__ out, const float* __restrict__ s1,
            const float* __restrict__ s2, const float* __restrict__ s3) {
    size_t i = ((size_t)blockIdx.x * 256 + threadIdx.x) * 4;
    if (i >= (size_t)Bn * Cc * Hh * Ww) return;
    int w = i & (Ww - 1);
    int h = (i >> 9) & (Hh - 1);
    int c = (i >> 14) & (Cc - 1);
    int b = i >> 21;
    float4 yv = *(const float4*)(out + i);
    float4 a1 = *(const float4*)(s1 + ((size_t)b * Hh + h) * Ww + w);
    float4 a2 = *(const float4*)(s2 + ((size_t)b * Cc + c) * Ww + w);
    float g3  = s3[((size_t)b * Hh + h) * Cc + c];
    const float k = 1.f / 3.f;
    yv.x = yv.x * (a1.x + a2.x + g3) * k;
    yv.y = yv.y * (a1.y + a2.y + g3) * k;
    yv.z = yv.z * (a1.z + a2.z + g3) * k;
    yv.w = yv.w * (a1.w + a2.w + g3) * k;
    *(float4*)(out + i) = yv;
}

extern "C" void kernel_launch(void* const* d_in, const int* in_sizes, int n_in,
                              void* d_out, int out_size, void* d_ws, size_t ws_size,
                              hipStream_t stream) {
    const float* x  = (const float*)d_in[0];
    const float* wg = (const float*)d_in[1];
    const float* w1 = (const float*)d_in[2];
    const float* w2 = (const float*)d_in[3];
    const float* w3 = (const float*)d_in[4];
    float* y = (float*)d_out;          // grouped-conv result lives in d_out

    float* ws = (float*)d_ws;
    __bf16* wbf = (__bf16*)ws;         // [3][128][64] bf16 = 12288 float slots
    float* p1 = ws + 12288;            // [B][2][H][W]  = 524288
    float* p2 = p1 + Bn * 2 * Hh * Ww; // [B][2][C][W]  = 2097152
    float* p3 = p2 + Bn * 2 * Cc * Ww; // [B][2][H][C]  = 131072
    float* s1 = p3 + Bn * 2 * Hh * Cc; // [B][H][W]     = 262144
    float* s2 = s1 + Bn * Hh * Ww;     // [B][C][W]     = 1048576
    float* s3 = s2 + Bn * Cc * Ww;     // [B][H][C]     = 65536

    // 0) one-time weight convert + reorder (bf16, fragment-friendly layout)
    prep_w<<<(NWELEM + 255) / 256, 256, 0, stream>>>(wg, wbf);

    // 1) grouped conv via WMMA
    conv_wmma<<<dim3(Ww / 64, Hh, Bn), 256, 0, stream>>>(x, wbf, y);

    // 2) pool3 init + fused triple pooling
    init_p3<<<(Bn * 2 * Hh * Cc + 255) / 256, 256, 0, stream>>>(p3);
    pools<<<dim3(Ww / 32, Bn), 256, 0, stream>>>(y, p1, p2, p3);

    // 3) tiny 7x7 convs + sigmoid (branch-3 mean plane holds a sum: scale 1/W)
    conv7_sig<<<(Bn * Hh * Ww + 255) / 256, 256, 0, stream>>>(p1, w1, s1, Hh, Ww, 1.f);
    conv7_sig<<<(Bn * Cc * Ww + 255) / 256, 256, 0, stream>>>(p2, w2, s2, Cc, Ww, 1.f);
    conv7_sig<<<(Bn * Hh * Cc + 255) / 256, 256, 0, stream>>>(p3, w3, s3, Hh, Cc, 1.f / Ww);

    // 4) gate + average, in place on d_out
    finalk<<<(Bn * Cc * Hh * Ww / 4 + 255) / 256, 256, 0, stream>>>(y, s1, s2, s3);
}
